// RecNN_69114613728754
// MI455X (gfx1250) — matrix operand order repaired and checked by reading, not compiled, not run on hardware
//
#include <hip/hip_runtime.h>
#include <stdint.h>

// ---------------------------------------------------------------------------
// RecNN tree MLP for MI455X (gfx1250): bf16 WMMA (16x16x32, f32 accum).
// Weight operand staged via TDM (tensor_load_to_lds) into a triple-buffered,
// bank-conflict-free (padded) LDS tile; one workgroup barrier per K-step.
// B=64, D=11, F=256, H=512.
// ---------------------------------------------------------------------------

typedef __bf16 bf16_t;
typedef __attribute__((ext_vector_type(16))) __bf16 v16bf;
typedef __attribute__((ext_vector_type(8)))  __bf16 v8bf;
typedef __attribute__((ext_vector_type(8)))  float  v8f;
typedef __attribute__((ext_vector_type(4)))  float  f32x4;
typedef __attribute__((ext_vector_type(4)))  unsigned int u32x4;
typedef __attribute__((ext_vector_type(8)))  int    i32x8;
typedef __attribute__((ext_vector_type(4)))  int    i32x4;

#define WMMA_BF16(a, b, c) \
  __builtin_amdgcn_wmma_f32_16x16x32_bf16(false, (a), false, (b), (short)0, (c), false, false)

// LDS tile geometry: 512 rows x 32 bf16 cols, padded by TDM to 80 B/row.
#define TILE_ROW_ELTS 40                    // 32 data + 8 pad bf16 (80 bytes)
#define TILE_BYTES    (512 * TILE_ROW_ELTS * 2)  // 40 KB per buffer

// Native f32 -> bf16 convert (backend emits hardware cvt, packable).
__device__ __forceinline__ bf16_t f2bf(float f) { return (bf16_t)f; }

// 16x32 bf16 A-fragment from a bf16 row (contiguous K layout):
// lane half 0 -> K = {k0..k0+15}, half 1 -> {k0+16..k0+31}.
__device__ __forceinline__ v16bf loadA_bf16(const bf16_t* __restrict__ row, int k0, int half) {
  return *(const v16bf*)(row + k0 + 16 * half);
}

// ---------------------------------------------------------------------------
// Tensor Data Mover: stage a [512 rows x 32 cols] bf16 tile into LDS with
// 16 B padding after every 64 B (bank-conflict-free ds_load_b128 fragments).
// D# packing per CDNA5 ISA (2-D tensor, groups 2/3 = 0).
// ---------------------------------------------------------------------------
__device__ __forceinline__ void tdm_issue(u32x4 g0, i32x8 g1) {
  i32x4 z4 = {0, 0, 0, 0};
#if defined(__clang_major__) && (__clang_major__ >= 23)
  i32x8 z8 = {0, 0, 0, 0, 0, 0, 0, 0};
  __builtin_amdgcn_tensor_load_to_lds(g0, g1, z4, z4, z8, 0);
#else
  __builtin_amdgcn_tensor_load_to_lds(g0, g1, z4, z4, 0);
#endif
}

__device__ __forceinline__ void tdm_load_tile(uint32_t ldsByteAddr,
                                              const bf16_t* tileStart,
                                              uint32_t rowLenElts) {
  uint64_t ga = (uint64_t)(uintptr_t)tileStart;
  u32x4 g0 = {0u, 0u, 0u, 0u};
  g0[0] = 1u;                                                   // count=1 (user)
  g0[1] = ldsByteAddr;                                          // LDS dest (bytes)
  g0[2] = (uint32_t)ga;                                         // global_addr[31:0]
  g0[3] = ((uint32_t)(ga >> 32) & 0x01FFFFFFu) | (2u << 30);    // [56:32] | type=2
  i32x8 g1 = {0, 0, 0, 0, 0, 0, 0, 0};
  g1[0] = (int)((1u << 16)      // data_size = 2 bytes
              | (1u << 20)      // pad_enable
              | (3u << 22)      // pad_interval: every 16 DWORDs (64 B)
              | (3u << 25));    // pad_amount: 4 DWORDs (16 B)
  g1[1] = (int)((rowLenElts & 0xFFFFu) << 16);                  // tensor_dim0[15:0]
  g1[2] = (int)(((rowLenElts >> 16) & 0xFFFFu) | (512u << 16)); // dim0[31:16]|dim1=512
  g1[3] = (int)(32u << 16);                                     // tile_dim0 = 32
  g1[4] = (int)(512u);                                          // tile_dim1 = 512
  g1[5] = (int)rowLenElts;                                      // tensor_dim0_stride
  tdm_issue(g0, g1);
}

// ---------------------------------------------------------------------------
// fp32 -> bf16 conversion, 8 elements/thread (for contents, W_u, W_h)
// ---------------------------------------------------------------------------
__global__ __launch_bounds__(256) void k_cvt_bf16(const float* __restrict__ src,
                                                  bf16_t* __restrict__ dst, int n8) {
  int i = blockIdx.x * 256 + threadIdx.x;
  if (i < n8) {
    f32x4 f0 = *(const f32x4*)(src + (size_t)i * 8);
    f32x4 f1 = *(const f32x4*)(src + (size_t)i * 8 + 4);
    v8bf o;
#pragma unroll
    for (int j = 0; j < 4; ++j) { o[j] = f2bf(f0[j]); o[4 + j] = f2bf(f1[j]); }
    *(v8bf*)(dst + (size_t)i * 8) = o;
  }
}

// ---------------------------------------------------------------------------
// Kernel 1: U = relu(X @ Wu^T + bu)   X:[M,256] bf16, Wu:[512,256] bf16
// WG: 64M x 512N, 8 waves (wave tile 32M x 128N). Wu staged via TDM.
// ---------------------------------------------------------------------------
__global__ __launch_bounds__(256) void k_gemm_u(const bf16_t* __restrict__ X,
                                                const bf16_t* __restrict__ Wu,
                                                const float* __restrict__ bu,
                                                bf16_t* __restrict__ U) {
  __shared__ __align__(128) bf16_t smB[3][512 * TILE_ROW_ELTS];   // 3 x 40 KB

  const int tid  = threadIdx.x;
  const int lane = tid & 31;
  const int wave = tid >> 5;
  const int wm   = wave & 1;
  const int wn   = wave >> 1;
  const int ln   = lane & 15;
  const int half = lane >> 4;
  const int nOff = wn * 128;

  const int mBlock = blockIdx.x * 64;
  const int m0 = mBlock + wm * 32 + ln;
  const bf16_t* rA0 = X + (size_t)m0 * 256;
  const bf16_t* rA1 = X + (size_t)(m0 + 16) * 256;

  v8f acc[2][8];
  v8f zero = {0.f, 0.f, 0.f, 0.f, 0.f, 0.f, 0.f, 0.f};
#pragma unroll
  for (int s = 0; s < 2; ++s)
#pragma unroll
    for (int t = 0; t < 8; ++t) acc[s][t] = zero;

  const uint32_t ldsBase = (uint32_t)(size_t)&smB[0][0];
  const int STEPS = 8;
  if (wave == 0) {
    tdm_load_tile(ldsBase,              Wu,      256u);
    tdm_load_tile(ldsBase + TILE_BYTES, Wu + 32, 256u);
  }

#pragma unroll 1
  for (int kt = 0; kt < STEPS; ++kt) {
    if (wave == 0) {
      if (kt + 1 < STEPS) __builtin_amdgcn_s_wait_tensorcnt(1);  // tile kt done
      else                __builtin_amdgcn_s_wait_tensorcnt(0);
    }
    __syncthreads();                       // tile[kt%3] visible; frees tile[(kt+2)%3]
    if (wave == 0 && kt + 2 < STEPS)
      tdm_load_tile(ldsBase + (uint32_t)((kt + 2) % 3) * TILE_BYTES,
                    Wu + (kt + 2) * 32, 256u);

    const bf16_t* sm = smB[kt % 3];
    const int k0 = kt * 32;
    v16bf a0 = loadA_bf16(rA0, k0, half);
    v16bf a1 = loadA_bf16(rA1, k0, half);

    v16bf bf[8];
#pragma unroll
    for (int t = 0; t < 8; ++t)
      bf[t] = *(const v16bf*)(sm + (nOff + t * 16 + ln) * TILE_ROW_ELTS + 16 * half);
#pragma unroll
    for (int t = 0; t < 8; ++t) {
      acc[0][t] = WMMA_BF16(a0, bf[t], acc[0][t]);
      acc[1][t] = WMMA_BF16(a1, bf[t], acc[1][t]);
    }
  }

#pragma unroll
  for (int s = 0; s < 2; ++s)
#pragma unroll
    for (int t = 0; t < 8; ++t) {
      const int col  = nOff + t * 16 + ln;
      const float bv = bu[col];
      const int row0 = mBlock + wm * 32 + s * 16 + 8 * half;
#pragma unroll
      for (int r = 0; r < 8; ++r) {
        float v = acc[s][t][r] + bv;
        v = v > 0.f ? v : 0.f;
        U[(size_t)(row0 + r) * 512 + col] = f2bf(v);
      }
    }
}

// ---------------------------------------------------------------------------
// Kernel 2: one tree level, fused gather-GEMM.
// embOut[m] = relu(concat(embPrev[cL], embPrev[cR], U[o+m]) @ Wh^T + bh)
// K = 1536 (48 steps of 32); Wh slice staged via TDM each step.
// ---------------------------------------------------------------------------
__global__ __launch_bounds__(256) void k_gemm_tree(const bf16_t* __restrict__ embPrev,
                                                   const bf16_t* __restrict__ U,
                                                   const int* __restrict__ children,
                                                   const bf16_t* __restrict__ Wh,
                                                   const float* __restrict__ bh,
                                                   int levelOff,
                                                   bf16_t* __restrict__ embOut,
                                                   float* __restrict__ outF) {
  __shared__ __align__(128) bf16_t smB[3][512 * TILE_ROW_ELTS];   // 3 x 40 KB

  const int tid  = threadIdx.x;
  const int lane = tid & 31;
  const int wave = tid >> 5;
  const int wm   = wave & 1;
  const int wn   = wave >> 1;
  const int ln   = lane & 15;
  const int half = lane >> 4;
  const int nOff = wn * 128;

  const int mBlock = blockIdx.x * 64;
  const int m0 = mBlock + wm * 32 + ln;
  const int m1 = m0 + 16;
  const int n0 = levelOff + m0;
  const int n1 = levelOff + m1;

  const bf16_t* rL0 = embPrev + (size_t)children[2 * n0]     * 512;
  const bf16_t* rR0 = embPrev + (size_t)children[2 * n0 + 1] * 512;
  const bf16_t* rU0 = U       + (size_t)n0 * 512;
  const bf16_t* rL1 = embPrev + (size_t)children[2 * n1]     * 512;
  const bf16_t* rR1 = embPrev + (size_t)children[2 * n1 + 1] * 512;
  const bf16_t* rU1 = U       + (size_t)n1 * 512;

  v8f acc[2][8];
  v8f zero = {0.f, 0.f, 0.f, 0.f, 0.f, 0.f, 0.f, 0.f};
#pragma unroll
  for (int s = 0; s < 2; ++s)
#pragma unroll
    for (int t = 0; t < 8; ++t) acc[s][t] = zero;

  const uint32_t ldsBase = (uint32_t)(size_t)&smB[0][0];
  const int STEPS = 48;
  if (wave == 0) {
    tdm_load_tile(ldsBase,              Wh,      1536u);
    tdm_load_tile(ldsBase + TILE_BYTES, Wh + 32, 1536u);
  }

#pragma unroll 1
  for (int kt = 0; kt < STEPS; ++kt) {
    if (wave == 0) {
      if (kt + 1 < STEPS) __builtin_amdgcn_s_wait_tensorcnt(1);
      else                __builtin_amdgcn_s_wait_tensorcnt(0);
    }
    __syncthreads();
    if (wave == 0 && kt + 2 < STEPS)
      tdm_load_tile(ldsBase + (uint32_t)((kt + 2) % 3) * TILE_BYTES,
                    Wh + (kt + 2) * 32, 1536u);

    const int seg = kt >> 4;                    // 0: h_L, 1: h_R, 2: u_j
    const int kk  = (kt & 15) * 32;
    const bf16_t* a0row = (seg == 0) ? rL0 : (seg == 1) ? rR0 : rU0;
    const bf16_t* a1row = (seg == 0) ? rL1 : (seg == 1) ? rR1 : rU1;
    v16bf a0 = loadA_bf16(a0row, kk, half);
    v16bf a1 = loadA_bf16(a1row, kk, half);

    const bf16_t* sm = smB[kt % 3];
    v16bf bf[8];
#pragma unroll
    for (int t = 0; t < 8; ++t)
      bf[t] = *(const v16bf*)(sm + (nOff + t * 16 + ln) * TILE_ROW_ELTS + 16 * half);
#pragma unroll
    for (int t = 0; t < 8; ++t) {
      acc[0][t] = WMMA_BF16(a0, bf[t], acc[0][t]);
      acc[1][t] = WMMA_BF16(a1, bf[t], acc[1][t]);
    }
  }

#pragma unroll
  for (int s = 0; s < 2; ++s)
#pragma unroll
    for (int t = 0; t < 8; ++t) {
      const int col  = nOff + t * 16 + ln;
      const float bv = bh[col];
      const int row0 = mBlock + wm * 32 + s * 16 + 8 * half;
#pragma unroll
      for (int r = 0; r < 8; ++r) {
        float v = acc[s][t][r] + bv;
        v = v > 0.f ? v : 0.f;
        const size_t idx = (size_t)(row0 + r) * 512 + col;
        embOut[idx] = f2bf(v);
        if (outF) outF[idx] = v;
      }
    }
}

// ---------------------------------------------------------------------------
// Host launcher
// ---------------------------------------------------------------------------
extern "C" void kernel_launch(void* const* d_in, const int* in_sizes, int n_in,
                              void* d_out, int out_size, void* d_ws, size_t ws_size,
                              hipStream_t stream) {
  (void)in_sizes; (void)n_in; (void)out_size; (void)ws_size;

  const int B = 64, D = 11, F = 256, H = 512;
  const int Ntot = B * ((1 << D) - 1);        // 131008 nodes

  const float* contents = (const float*)d_in[0];   // [Ntot, F]
  const int*   children = (const int*)d_in[1];     // [65472, 2]
  const float* W_u      = (const float*)d_in[2];   // [H, F]
  const float* b_u      = (const float*)d_in[3];   // [H]
  const float* W_h      = (const float*)d_in[4];   // [H, 3H]
  const float* b_h      = (const float*)d_in[5];   // [H]
  float*       out      = (float*)d_out;           // [B, H]

  // Workspace carve-out (all offsets 32B aligned).
  uint8_t* ws = (uint8_t*)d_ws;
  size_t off = 0;
  bf16_t* Ubf = (bf16_t*)(ws + off); off += (size_t)Ntot * H * sizeof(bf16_t);      // 134 MB
  bf16_t* Xbf = (bf16_t*)(ws + off); off += (size_t)Ntot * F * sizeof(bf16_t);      // 67 MB
  bf16_t* Wu  = (bf16_t*)(ws + off); off += (size_t)H * F * sizeof(bf16_t);         // 256 KB
  bf16_t* Wh  = (bf16_t*)(ws + off); off += (size_t)H * 3 * H * sizeof(bf16_t);     // 1.5 MB
  bf16_t* embA = (bf16_t*)(ws + off); off += (size_t)(B << (D - 2)) * H * sizeof(bf16_t); // 32 MB
  bf16_t* embB = (bf16_t*)(ws + off);

  // 1) Convert inputs/weights to bf16 (vectorized, 8 elems/thread).
  {
    int nX = Ntot * F / 8;
    k_cvt_bf16<<<(nX + 255) / 256, 256, 0, stream>>>(contents, Xbf, nX);
    int nWu = H * F / 8;
    k_cvt_bf16<<<(nWu + 255) / 256, 256, 0, stream>>>(W_u, Wu, nWu);
    int nWh = H * 3 * H / 8;
    k_cvt_bf16<<<(nWh + 255) / 256, 256, 0, stream>>>(W_h, Wh, nWh);
  }

  // 2) u = relu(contents @ Wu^T + bu) for all nodes.
  k_gemm_u<<<Ntot / 64, 256, 0, stream>>>(Xbf, Wu, b_u, Ubf);

  // 3) Tree reduction: leaves (level D-1, u rows) up to root level 0.
  const bf16_t* embPrev = Ubf + (size_t)(B * ((1 << (D - 1)) - 1)) * H; // u[level 10]
  bf16_t* bufs[2] = {embA, embB};
  int pp = 0;
  for (int j = D - 2; j >= 0; --j) {
    const int n = B << j;
    const int o = B * ((1 << j) - 1);
    bf16_t* eOut = bufs[pp];
    float*  fOut = (j == 0) ? out : nullptr;
    k_gemm_tree<<<n / 64, 256, 0, stream>>>(embPrev, Ubf, children, Wh, b_h,
                                            o, eOut, fOut);
    embPrev = eOut;
    pp ^= 1;
  }
}
